// EfficientTransformer_57002805952728
// MI455X (gfx1250) — compile-verified, matrix operand order checked by reading, not run on hardware
//
#include <hip/hip_runtime.h>
#include <hip/hip_bf16.h>

typedef __attribute__((ext_vector_type(16))) __bf16    v16bf;
typedef __attribute__((ext_vector_type(2)))  __bf16    v2bf;
typedef __attribute__((ext_vector_type(8)))  float     v8f;
typedef __attribute__((ext_vector_type(8)))  unsigned  v8u;
typedef __attribute__((ext_vector_type(4)))  unsigned  v4u;
typedef __attribute__((ext_vector_type(2)))  unsigned  v2u;
typedef __attribute__((ext_vector_type(4)))  float     v4f;
typedef __attribute__((ext_vector_type(2)))  float     v2f;

#define B_   4
#define S_   8192
#define F_   32
#define D_   512
#define H_   8
#define HD_  64
#define L_   4
#define FD_  128
#define WIN_ 256
#define CF_  4

// padded row strides (u16 units); pad multiple of 8 u16 keeps 16B alignment
#define AROW_ 40     // 32 k + 8 pad       (20 dwords)
#define BROW_ 40     // 32 k + 8 pad       (20 dwords)
#define KROW_ 72     // 64 d + 8 pad       (36 dwords)
#define VROW_ 536    // 512 keys + 24 pad  (268 dwords)
#define PROW_ 40     // 32 n + 8 pad       (20 dwords)

__device__ __forceinline__ unsigned short f32_to_bf16(float f) {
    return __builtin_bit_cast(unsigned short, (__bf16)f);
}
// one v_cvt_pk_bf16_f32: two f32 -> packed bf16x2 dword
__device__ __forceinline__ unsigned pack2_bf16(v2f t) {
    v2bf h = __builtin_convertvector(t, v2bf);
    return __builtin_bit_cast(unsigned, h);
}
__device__ __forceinline__ unsigned pack_bf16(float lo, float hi) {
    v2f t; t[0] = lo; t[1] = hi;
    return pack2_bf16(t);
}

// Fragment = 8 dwords per lane: dwords 0..3 at base, 4..7 at base+8 (two b128)
__device__ __forceinline__ v16bf load_frag(const unsigned* p) {
    v4u lo = *(const v4u*)p;
    v4u hi = *(const v4u*)(p + 8);
    v8u f = __builtin_shufflevector(lo, hi, 0, 1, 2, 3, 4, 5, 6, 7);
    return __builtin_bit_cast(v16bf, f);
}

// ---------------------------------------------------------------------------
// bf16-WMMA GEMM:  C[M,N] = act( A[M,K] @ W[K,N] + bias ),  A optionally split
// Block tile 256x128, 256 threads (8 waves): wave = 2 strips x 16 rows, 128 N.
// Requirements (all call sites satisfy): M%256==0, K%32==0, K0%32==0.
// Only N may be ragged (final GEMM N=32).
// Register-prefetch pipeline: fetch tile k+1 while computing tile k.
// ---------------------------------------------------------------------------
__global__ void __launch_bounds__(256)
gemm_bf16_kernel(const float* __restrict__ A0, int K0, const float* __restrict__ A1,
                 const float* __restrict__ W, const float* __restrict__ bias,
                 float* __restrict__ C, int M, int N, int K, int act)
{
    __shared__ unsigned short Araw[256 * AROW_];   // 20 KB  [m][k]
    __shared__ unsigned short Braw[128 * BROW_];   // 10 KB  transposed [n][k]

    const int tid  = threadIdx.x;
    const int wave = tid >> 5;
    const int lane = tid & 31;
    const int hl   = lane & 15;
    const int kb2  = (lane < 16) ? 0 : 4;
    const int m0   = blockIdx.y * 256;
    const int n0   = blockIdx.x * 128;
    const int K1   = K - K0;
    const bool nfull = (n0 + 128 <= N);

    const unsigned* A32 = (const unsigned*)Araw;
    const unsigned* B32 = (const unsigned*)Braw;

    v4f ra[8];   // A prefetch regs: j -> row (j*256+tid)>>3, col ((j*256+tid)&7)*4
    v4f rb[4];   // B prefetch regs: j -> row (j*256+tid)>>5, col ((j*256+tid)&31)*4

    auto fetch = [&](int k0) {
        const bool  useA0 = (k0 < K0);
        const float* Ab   = useA0 ? A0 : A1;
        const int    ld   = useA0 ? K0 : K1;
        const int    ko   = useA0 ? k0 : k0 - K0;
#pragma unroll
        for (int j = 0; j < 8; ++j) {
            int f = j * 256 + tid;
            int r = f >> 3, c = (f & 7) * 4;
            ra[j] = *(const v4f*)(Ab + (long)(m0 + r) * ld + ko + c);
        }
#pragma unroll
        for (int j = 0; j < 4; ++j) {
            int f = j * 256 + tid;
            int r = f >> 5, c = (f & 31) * 4;
            if (nfull) {
                rb[j] = *(const v4f*)(W + (long)(k0 + r) * N + n0 + c);
            } else {
                v4f t = {};
#pragma unroll
                for (int e = 0; e < 4; ++e)
                    if (n0 + c + e < N) t[e] = W[(long)(k0 + r) * N + n0 + c + e];
                rb[j] = t;
            }
        }
    };

    auto stage = [&]() {
#pragma unroll
        for (int j = 0; j < 8; ++j) {
            int f = j * 256 + tid;
            int r = f >> 3, c = (f & 7) * 4;
            v2u p = { pack2_bf16(__builtin_shufflevector(ra[j], ra[j], 0, 1)),
                      pack2_bf16(__builtin_shufflevector(ra[j], ra[j], 2, 3)) };
            *(v2u*)&Araw[r * AROW_ + c] = p;      // 8B-aligned b64 store
        }
#pragma unroll
        for (int j = 0; j < 4; ++j) {
            int f = j * 256 + tid;
            int r = f >> 5, c = (f & 31) * 4;
#pragma unroll
            for (int e = 0; e < 4; ++e)
                Braw[(c + e) * BROW_ + r] = f32_to_bf16(rb[j][e]);
        }
    };

    v8f acc[2][8] = {};

    fetch(0);
    for (int k0 = 0; k0 < K; k0 += 32) {
        stage();
        __syncthreads();
        if (k0 + 32 < K) fetch(k0 + 32);    // overlap next-tile loads with wmma

        v16bf af[2];
#pragma unroll
        for (int s = 0; s < 2; ++s) {
            int rl = (wave * 2 + s) * 16 + hl;
            af[s] = load_frag(A32 + rl * (AROW_ / 2) + kb2);
        }
#pragma unroll
        for (int nt = 0; nt < 8; ++nt) {
            v16bf bf = load_frag(B32 + (nt * 16 + hl) * (BROW_ / 2) + kb2);
            acc[0][nt] = __builtin_amdgcn_wmma_f32_16x16x32_bf16(
                false, af[0], false, bf, (short)0, acc[0][nt], false, false);
            acc[1][nt] = __builtin_amdgcn_wmma_f32_16x16x32_bf16(
                false, af[1], false, bf, (short)0, acc[1][nt], false, false);
        }
        __syncthreads();
    }

    // C layout: (vgpr r, lane) -> M = r + 8*(lane>>4), N = lane&15
#pragma unroll
    for (int s = 0; s < 2; ++s) {
        const int rbase = m0 + (wave * 2 + s) * 16 + ((lane >> 4) << 3);
#pragma unroll
        for (int nt = 0; nt < 8; ++nt) {
            int gn = n0 + nt * 16 + hl;
            if (gn >= N) continue;
            float bv = bias ? bias[gn] : 0.f;
#pragma unroll
            for (int r = 0; r < 8; ++r) {
                float val = acc[s][nt][r] + bv;
                if (act == 1) val = 1.f / (1.f + __expf(-val));
                C[(long)(rbase + r) * N + gn] = val;
            }
        }
    }
}

// ---------------------------------------------------------------------------
// Sliding-window flash attention. Grid = (nb, H, B), 512 threads (16 waves).
// ---------------------------------------------------------------------------
__global__ void __launch_bounds__(512)
attn_kernel(const float* __restrict__ q, const float* __restrict__ k,
            const float* __restrict__ v, float* __restrict__ o)
{
    extern __shared__ unsigned short smem[];
    unsigned short* Kraw = smem;                            // 512*KROW_
    unsigned short* Vt   = smem + 512 * KROW_;              // 64*VROW_ (transposed)
    unsigned short* Praw = smem + 512 * KROW_ + 64 * VROW_; // 16 waves * 16*PROW_
    const unsigned* K32 = (const unsigned*)Kraw;
    const unsigned* V32 = (const unsigned*)Vt;
    unsigned* K32w = (unsigned*)Kraw;

    const int blk = blockIdx.x, hh = blockIdx.y, b = blockIdx.z;
    const int tid = threadIdx.x, wave = tid >> 5, lane = tid & 31;
    const int hl  = lane & 15;
    const int kb2 = (lane < 16) ? 0 : 4;
    const int kbL = kb2 * 2;
    const int kstart = (blk == 0) ? WIN_ : 0;

    // ---- stage K (row-major padded) + V (transposed padded), guard hoisted --
    const long kvbase = (((long)b * S_) + (long)(blk - 1) * WIN_) * D_ + hh * HD_;
    for (int i = tid; i < kstart * 32; i += 512) {          // zero prev-window
        int kidx = i >> 5, d2 = i & 31, d = d2 * 2;
        K32w[kidx * (KROW_ / 2) + d2] = 0u;
        Vt[d * VROW_ + kidx] = 0;
        Vt[(d + 1) * VROW_ + kidx] = 0;
    }
    for (int i = kstart * 32 + tid; i < 512 * 32; i += 512) {
        int kidx = i >> 5, d2 = i & 31, d = d2 * 2;
        long g = kvbase + (long)kidx * D_ + d;
        v2f kf = *(const v2f*)(k + g);
        v2f vf = *(const v2f*)(v + g);
        K32w[kidx * (KROW_ / 2) + d2] = pack2_bf16(kf);
        Vt[d * VROW_ + kidx]       = f32_to_bf16(vf[0]);
        Vt[(d + 1) * VROW_ + kidx] = f32_to_bf16(vf[1]);
    }
    __syncthreads();

    // ---- Q fragments (one-time, direct from global) ------------------------
    const long qbase = (((long)b * S_) + blk * WIN_ + wave * 16 + hl) * D_ + hh * HD_;
    v16bf aq[2];
#pragma unroll
    for (int hf = 0; hf < 2; ++hf) {
        v8u t;
#pragma unroll
        for (int i = 0; i < 8; ++i) {
            int kk = kbL + 2 * i + ((i & 4) << 1);
            v2f qq = *(const v2f*)(q + qbase + hf * 32 + kk);
            t[i] = pack2_bf16(qq);
        }
        aq[hf] = __builtin_bit_cast(v16bf, t);
    }

    v8f oacc[4] = {};
    float mrun[8], lrun[8];
#pragma unroll
    for (int r = 0; r < 8; ++r) { mrun[r] = -1e30f; lrun[r] = 0.f; }

    unsigned short* Pw  = Praw + wave * 16 * PROW_;
    const unsigned* P32 = (const unsigned*)Pw;
    const float scale = 0.125f;   // 1/sqrt(64)

    for (int kc = kstart; kc < 2 * WIN_; kc += 32) {
        // ---- scores: two 16x16 C tiles over HD=64 (2 wmma each) -----------
        v8f s0 = {}, s1 = {};
#pragma unroll
        for (int hf = 0; hf < 2; ++hf) {
            v16bf b0 = load_frag(K32 + (kc + hl) * (KROW_ / 2) + hf * 16 + kb2);
            v16bf b1 = load_frag(K32 + (kc + 16 + hl) * (KROW_ / 2) + hf * 16 + kb2);
            s0 = __builtin_amdgcn_wmma_f32_16x16x32_bf16(false, aq[hf], false, b0, (short)0, s0, false, false);
            s1 = __builtin_amdgcn_wmma_f32_16x16x32_bf16(false, aq[hf], false, b1, (short)0, s1, false, false);
        }

        // ---- online softmax over these 32 keys ----------------------------
#pragma unroll
        for (int r = 0; r < 8; ++r) {
            float x0 = s0[r] * scale, x1 = s1[r] * scale;
            float mx = fmaxf(x0, x1);
#pragma unroll
            for (int off = 1; off < 16; off <<= 1)
                mx = fmaxf(mx, __shfl_xor(mx, off, 32));
            float mnew = fmaxf(mrun[r], mx);
            float corr = __expf(mrun[r] - mnew);
            float p0 = __expf(x0 - mnew), p1 = __expf(x1 - mnew);
            float ps = p0 + p1;
#pragma unroll
            for (int off = 1; off < 16; off <<= 1)
                ps += __shfl_xor(ps, off, 32);
            lrun[r] = lrun[r] * corr + ps;
            mrun[r] = mnew;
#pragma unroll
            for (int nt = 0; nt < 4; ++nt) oacc[nt][r] *= corr;
            int prow = r + ((lane >> 4) << 3);
            Pw[prow * PROW_ + hl]      = f32_to_bf16(p0);
            Pw[prow * PROW_ + 16 + hl] = f32_to_bf16(p1);
        }

        // ---- P as A fragment (two b128 from per-wave patch) ---------------
        v16bf pa = load_frag(P32 + hl * (PROW_ / 2) + kb2);

        // ---- O += P @ V[kc:kc+32, :] --------------------------------------
#pragma unroll
        for (int nt = 0; nt < 4; ++nt) {
            v16bf bv = load_frag(V32 + (nt * 16 + hl) * (VROW_ / 2) + (kc >> 1) + kb2);
            oacc[nt] = __builtin_amdgcn_wmma_f32_16x16x32_bf16(false, pa, false, bv, (short)0, oacc[nt], false, false);
        }
    }

    // ---- normalize + store -------------------------------------------------
#pragma unroll
    for (int nt = 0; nt < 4; ++nt) {
#pragma unroll
        for (int r = 0; r < 8; ++r) {
            long row = (long)b * S_ + blk * WIN_ + wave * 16 + r + ((lane >> 4) << 3);
            o[row * D_ + hh * HD_ + nt * 16 + hl] = oacc[nt][r] / lrun[r];
        }
    }
}

// ---------------------------------------------------------------------------
__global__ void __launch_bounds__(256)
gather_te_kernel(const int* __restrict__ ts, const float* __restrict__ emb,
                 float* __restrict__ te, int M)
{
    long idx = (long)blockIdx.x * 256 + threadIdx.x;
    if (idx >= (long)M * FD_) return;
    long m = idx >> 7;
    int  d = idx & 127;
    te[idx] = emb[(long)ts[m] * FD_ + d];
}

// conv_w [D,D,CF] -> Wc [CF*D, D] with Wc[j*D+c, o] = conv_w[o,c,j]
__global__ void __launch_bounds__(256)
repack_conv_kernel(const float* __restrict__ cw, float* __restrict__ Wc)
{
    long idx = (long)blockIdx.x * 256 + threadIdx.x;
    if (idx >= (long)CF_ * D_ * D_) return;
    long kk = idx >> 9;
    int  o  = idx & 511;
    int  j  = (int)(kk >> 9);
    int  c  = (int)(kk & 511);
    Wc[idx] = cw[((long)o * D_ + c) * CF_ + j];
}

// new_mem = g*u + (1-g)*mem ; h = LN(a + new_mem)   (one token per block)
__global__ void __launch_bounds__(128)
gate_mem_ln_kernel(const float* __restrict__ a, const float* __restrict__ g,
                   const float* __restrict__ u, float* __restrict__ mem,
                   const float* __restrict__ gamma, const float* __restrict__ beta,
                   float* __restrict__ out)
{
    long t = blockIdx.x;
    int tid = threadIdx.x, lane = tid & 31, wave = tid >> 5;
    __shared__ float red[2][4];
    float xs[4];
#pragma unroll
    for (int i = 0; i < 4; ++i) {
        int d = tid + i * 128;
        long idx = t * D_ + d;
        float gg = g[idx], uu = u[idx], mm = mem[idx];
        float nm = gg * uu + (1.f - gg) * mm;
        mem[idx] = nm;
        xs[i] = a[idx] + nm;
    }
    float s = xs[0] + xs[1] + xs[2] + xs[3];
    float s2 = xs[0]*xs[0] + xs[1]*xs[1] + xs[2]*xs[2] + xs[3]*xs[3];
#pragma unroll
    for (int off = 1; off < 32; off <<= 1) {
        s  += __shfl_xor(s,  off, 32);
        s2 += __shfl_xor(s2, off, 32);
    }
    if (lane == 0) { red[0][wave] = s; red[1][wave] = s2; }
    __syncthreads();
    float S1 = red[0][0] + red[0][1] + red[0][2] + red[0][3];
    float S2 = red[1][0] + red[1][1] + red[1][2] + red[1][3];
    float mean = S1 * (1.f / D_);
    float var  = S2 * (1.f / D_) - mean * mean;
    float inv  = rsqrtf(var + 1e-5f);
#pragma unroll
    for (int i = 0; i < 4; ++i) {
        int d = tid + i * 128;
        out[t * D_ + d] = (xs[i] - mean) * inv * gamma[d] + beta[d];
    }
}

// c = LN(LN(c; g1,b1); g2,b2)   in place, one token per block
__global__ void __launch_bounds__(128)
double_ln_kernel(float* __restrict__ c, const float* __restrict__ g1,
                 const float* __restrict__ b1, const float* __restrict__ g2,
                 const float* __restrict__ b2)
{
    long t = blockIdx.x;
    int tid = threadIdx.x, lane = tid & 31, wave = tid >> 5;
    __shared__ float red[2][4];
    float xs[4];
#pragma unroll
    for (int i = 0; i < 4; ++i) xs[i] = c[t * D_ + tid + i * 128];

    for (int pass = 0; pass < 2; ++pass) {
        float s = xs[0] + xs[1] + xs[2] + xs[3];
        float s2 = xs[0]*xs[0] + xs[1]*xs[1] + xs[2]*xs[2] + xs[3]*xs[3];
#pragma unroll
        for (int off = 1; off < 32; off <<= 1) {
            s  += __shfl_xor(s,  off, 32);
            s2 += __shfl_xor(s2, off, 32);
        }
        if (lane == 0) { red[0][wave] = s; red[1][wave] = s2; }
        __syncthreads();
        float S1 = red[0][0] + red[0][1] + red[0][2] + red[0][3];
        float S2 = red[1][0] + red[1][1] + red[1][2] + red[1][3];
        float mean = S1 * (1.f / D_);
        float var  = S2 * (1.f / D_) - mean * mean;
        float inv  = rsqrtf(var + 1e-5f);
        const float* g = pass ? g2 : g1;
        const float* b = pass ? b2 : b1;
#pragma unroll
        for (int i = 0; i < 4; ++i) {
            int d = tid + i * 128;
            xs[i] = (xs[i] - mean) * inv * g[d] + b[d];
        }
        __syncthreads();
    }
#pragma unroll
    for (int i = 0; i < 4; ++i) c[t * D_ + tid + i * 128] = xs[i];
}

// ---------------------------------------------------------------------------
extern "C" void kernel_launch(void* const* d_in, const int* in_sizes, int n_in,
                              void* d_out, int out_size, void* d_ws, size_t ws_size,
                              hipStream_t stream)
{
    const float* x        = (const float*)d_in[0];
    const int*   ts       = (const int*)  d_in[1];
    const float* W_feat   = (const float*)d_in[2];
    const float* b_feat   = (const float*)d_in[3];
    const float* emb_temp = (const float*)d_in[4];
    const float* W_emb    = (const float*)d_in[5];
    const float* b_emb    = (const float*)d_in[6];
    const float* Wq = (const float*)d_in[7];   const float* bq = (const float*)d_in[8];
    const float* Wk = (const float*)d_in[9];   const float* bk = (const float*)d_in[10];
    const float* Wv = (const float*)d_in[11];  const float* bv = (const float*)d_in[12];
    const float* Wo = (const float*)d_in[13];  const float* bo = (const float*)d_in[14];
    const float* Wg = (const float*)d_in[15];  const float* bg = (const float*)d_in[16];
    const float* Wu = (const float*)d_in[17];  const float* bu = (const float*)d_in[18];
    const float* ln_g = (const float*)d_in[19]; const float* ln_b = (const float*)d_in[20];
    const float* conv_w = (const float*)d_in[21]; const float* conv_b = (const float*)d_in[22];
    const float* cn_g = (const float*)d_in[23]; const float* cn_b = (const float*)d_in[24];
    const float* on_g = (const float*)d_in[25]; const float* on_b = (const float*)d_in[26];
    const float* W_out = (const float*)d_in[27]; const float* b_out = (const float*)d_in[28];

    const int M = B_ * S_;               // 32768 tokens
    const size_t MB = 1024 * 1024;
    char* ws = (char*)d_ws;
    float* h    = (float*)(ws);              // 64 MB  [M, D]
    float* mem  = (float*)(ws + 64  * MB);   // 64 MB
    float* qb   = (float*)(ws + 128 * MB);   // 64 MB  (also fe / Wc)
    float* kbuf = (float*)(ws + 192 * MB);   // 64 MB  (also te / conv out)
    float* vbuf = (float*)(ws + 256 * MB);   // 64 MB  (also upd)
    float* atn  = (float*)(ws + 320 * MB);   // 64 MB  (also gate)
    float* abuf = (float*)(ws + 384 * MB);   // 64 MB
    float* fe   = qb;
    float* te   = kbuf;
    float* gate = atn;
    float* upd  = vbuf;
    float* Wc   = qb;
    float* cbuf = kbuf;

    auto gemm = [&](const float* A0, int K0, const float* A1, const float* W,
                    const float* bias, float* C, int m, int n, int kdim, int act) {
        dim3 grid((n + 127) / 128, (m + 255) / 256);
        gemm_bf16_kernel<<<grid, 256, 0, stream>>>(A0, K0, A1, W, bias, C, m, n, kdim, act);
    };

    // ---- embedding: h = [x@W_feat+b_feat, emb_temp[ts]] @ W_emb + b_emb ----
    gemm(x, F_, nullptr, W_feat, b_feat, fe, M, FD_, F_, 0);
    gather_te_kernel<<<((long)M * FD_ + 255) / 256, 256, 0, stream>>>(ts, emb_temp, te, M);
    gemm(fe, FD_, te, W_emb, b_emb, h, M, D_, 2 * FD_, 0);
    hipMemsetAsync(mem, 0, (size_t)M * D_ * sizeof(float), stream);

    // ---- layers ------------------------------------------------------------
    const size_t attn_lds = (size_t)(512 * KROW_ + 64 * VROW_ + 16 * 16 * PROW_)
                            * sizeof(unsigned short);   // ~163 KB
    for (int l = 0; l < L_; ++l) {
        const long DD = (long)D_ * D_;
        gemm(h, D_, nullptr, Wq + l * DD, bq + l * D_, qb,   M, D_, D_, 0);
        gemm(h, D_, nullptr, Wk + l * DD, bk + l * D_, kbuf, M, D_, D_, 0);
        gemm(h, D_, nullptr, Wv + l * DD, bv + l * D_, vbuf, M, D_, D_, 0);
        attn_kernel<<<dim3(S_ / WIN_, H_, B_), 512, attn_lds, stream>>>(qb, kbuf, vbuf, atn);
        gemm(atn, D_, nullptr, Wo + l * DD, bo + l * D_, abuf, M, D_, D_, 0);
        gemm(abuf, D_, mem, Wg + l * 2 * DD, bg + l * D_, gate, M, D_, 2 * D_, 1);
        gemm(abuf, D_, mem, Wu + l * 2 * DD, bu + l * D_, upd,  M, D_, 2 * D_, 0);
        gate_mem_ln_kernel<<<M, 128, 0, stream>>>(abuf, gate, upd, mem,
                                                  ln_g + l * D_, ln_b + l * D_, h);
    }

    // ---- temporal compression conv as GEMM [M/CF, CF*D] @ [CF*D, D] --------
    repack_conv_kernel<<<((long)CF_ * D_ * D_ + 255) / 256, 256, 0, stream>>>(conv_w, Wc);
    const int Mc = M / CF_;   // 8192
    gemm(h, CF_ * D_, nullptr, Wc, conv_b, cbuf, Mc, D_, CF_ * D_, 0);
    double_ln_kernel<<<Mc, 128, 0, stream>>>(cbuf, cn_g, cn_b, on_g, on_b);

    // ---- output projection -------------------------------------------------
    gemm(cbuf, D_, nullptr, W_out, b_out, (float*)d_out, Mc, F_, D_, 0);
}
